// LEDBasicSentenceClassificationModel_88227218194837
// MI455X (gfx1250) — matrix operand order repaired and checked by reading, not compiled, not run on hardware
//
#include <hip/hip_runtime.h>
#include <hip/hip_bf16.h>
#include <math.h>

// ---------------------------------------------------------------------------
// LED / Longformer encoder for MI455X (gfx1250, wave32, WMMA f16 -> f32,
// async-to-LDS double-buffered GEMM staging)
// ---------------------------------------------------------------------------

typedef __attribute__((ext_vector_type(16))) _Float16 v16h;
typedef __attribute__((ext_vector_type(8)))  _Float16 v8h;
typedef __attribute__((ext_vector_type(8)))  float    v8f;

#define LED_S   8192
#define LED_D   768
#define LED_H   12
#define LED_DH  64
#define LED_W   256
#define LED_C   32
#define LED_FF  3072
#define LED_N   128
#define LED_NEG (-1.0e9f)
#define LED_SCALE 0.125f   // 1/sqrt(64)

// Async copy: 16 bytes global -> LDS, tracked on ASYNCcnt.
// GVS mode: mem = SADDR(64b) + VADDR(i32 byte offset); LDS dst from VGPR.
#define LED_ASYNC_B128(ldsoff, goff, base)                        \
  asm volatile("global_load_async_to_lds_b128 %0, %1, %2"         \
               :: "v"(ldsoff), "v"(goff), "s"(base) : "memory")

__device__ __forceinline__ uint32_t led_lds_off(const void* p) {
  // low 32 bits of a flat LDS address == byte offset in the LDS aperture
  return (uint32_t)(uintptr_t)p;
}

// ---------------------------------------------------------------------------
// Fragment gathers from LDS matching the CDNA5 16-bit WMMA register layouts.
// ---------------------------------------------------------------------------
__device__ __forceinline__ v16h led_fragA(const _Float16* rowptr) {
  const int lane = threadIdx.x & 31;
  const int kb = (lane & 16) ? 8 : 0;
  v8h lo = *(const v8h*)(rowptr + kb);
  v8h hi = *(const v8h*)(rowptr + kb + 16);
  v16h r;
#pragma unroll
  for (int i = 0; i < 8; ++i) { r[i] = lo[i]; r[8 + i] = hi[i]; }
  return r;
}

__device__ __forceinline__ v16h led_fragB(const _Float16* colptr) {
  const int lane = threadIdx.x & 31;
  const int kb = (lane & 16) ? 8 : 0;
  v8h lo = *(const v8h*)(colptr + kb);
  v8h hi = *(const v8h*)(colptr + kb + 16);
  v16h r;
#pragma unroll
  for (int i = 0; i < 8; ++i) { r[i] = lo[i]; r[8 + i] = hi[i]; }
  return r;
}

__device__ __forceinline__ float led_block_sum768(float s, float* sred) {
#pragma unroll
  for (int off = 16; off > 0; off >>= 1) s += __shfl_xor(s, off, 32);
  const int wave = threadIdx.x >> 5;
  const int lane = threadIdx.x & 31;
  if (lane == 0) sred[wave] = s;
  __syncthreads();
  float tot = 0.f;
#pragma unroll
  for (int i = 0; i < 8; ++i) tot += sred[i];
  __syncthreads();
  return tot;
}

// ---------------------------------------------------------------------------
// Weight prep: W[K x N] f32  ->  WT[N x K] f16  (tiled LDS transpose)
// ---------------------------------------------------------------------------
__global__ __launch_bounds__(256) void led_w_to_f16t(
    const float* __restrict__ W, _Float16* __restrict__ WT, int K, int N) {
  __shared__ float tile[32][33];
  const int tid = threadIdx.x;
  const int n0 = blockIdx.x * 32, k0 = blockIdx.y * 32;
#pragma unroll
  for (int i = 0; i < 4; ++i) {
    const int idx = tid + i * 256;
    const int r = idx >> 5, c = idx & 31;           // r = k, c = n
    tile[r][c] = W[(size_t)(k0 + r) * N + (n0 + c)];
  }
  __syncthreads();
#pragma unroll
  for (int i = 0; i < 4; ++i) {
    const int idx = tid + i * 256;
    const int r = idx >> 5, c = idx & 31;           // r = n, c = k
    WT[(size_t)(n0 + r) * K + (k0 + c)] = (_Float16)tile[c][r];
  }
}

// ---------------------------------------------------------------------------
// Embedding + LayerNorm (writes f32 master + f16 mirror)
// ---------------------------------------------------------------------------
__global__ __launch_bounds__(256) void led_embed_ln(
    const int* __restrict__ src, const float* __restrict__ emb,
    const float* __restrict__ pos, const float* __restrict__ gs,
    const float* __restrict__ gb, float* __restrict__ out,
    _Float16* __restrict__ outh) {
  const int row = blockIdx.x;
  const int tid = threadIdx.x;
  __shared__ float sred[8];
  const int tok = src[row];
  float v[3];
#pragma unroll
  for (int i = 0; i < 3; ++i) {
    const int c = tid + i * 256;
    v[i] = emb[(size_t)tok * LED_D + c] + pos[(size_t)row * LED_D + c];
  }
  const float mean = led_block_sum768(v[0] + v[1] + v[2], sred) * (1.f / LED_D);
  const float d0 = v[0] - mean, d1 = v[1] - mean, d2 = v[2] - mean;
  const float var =
      led_block_sum768(d0 * d0 + d1 * d1 + d2 * d2, sred) * (1.f / LED_D);
  const float rs = rsqrtf(var + 1e-5f);
  const float dd[3] = {d0, d1, d2};
#pragma unroll
  for (int i = 0; i < 3; ++i) {
    const int c = tid + i * 256;
    const float o = dd[i] * rs * gs[c] + gb[c];
    out[(size_t)row * LED_D + c] = o;
    outh[(size_t)row * LED_D + c] = (_Float16)o;
  }
}

// ---------------------------------------------------------------------------
// Residual add + LayerNorm (writes f32 master + f16 mirror; in-place safe)
// ---------------------------------------------------------------------------
__global__ __launch_bounds__(256) void led_add_ln(
    const float* __restrict__ xin, const float* __restrict__ hin,
    const float* __restrict__ gs, const float* __restrict__ gb,
    float* __restrict__ out, _Float16* __restrict__ outh) {
  const int row = blockIdx.x;
  const int tid = threadIdx.x;
  __shared__ float sred[8];
  float v[3];
#pragma unroll
  for (int i = 0; i < 3; ++i) {
    const int c = tid + i * 256;
    v[i] = xin[(size_t)row * LED_D + c] + hin[(size_t)row * LED_D + c];
  }
  const float mean = led_block_sum768(v[0] + v[1] + v[2], sred) * (1.f / LED_D);
  const float d0 = v[0] - mean, d1 = v[1] - mean, d2 = v[2] - mean;
  const float var =
      led_block_sum768(d0 * d0 + d1 * d1 + d2 * d2, sred) * (1.f / LED_D);
  const float rs = rsqrtf(var + 1e-5f);
  const float dd[3] = {d0, d1, d2};
#pragma unroll
  for (int i = 0; i < 3; ++i) {
    const int c = tid + i * 256;
    const float o = dd[i] * rs * gs[c] + gb[c];
    out[(size_t)row * LED_D + c] = o;
    outh[(size_t)row * LED_D + c] = (_Float16)o;
  }
}

// ---------------------------------------------------------------------------
// WMMA GEMM: Y[M x Nn] = act( A[M x K](f16) @ WT[Nn x K](f16)^T + bias )
// 64x64 block tile, K-step 64, async-to-LDS double-buffered staging.
// 8 waves, each 16x32 output (2 accs, 4 wmma per K-step).
// act: 0=none, 1=gelu(exact), 2=tanh.  out16: write f16 (Yh) else f32 (Yf).
// ---------------------------------------------------------------------------
__global__ __launch_bounds__(256) void led_gemm_wmma(
    const _Float16* __restrict__ A, const _Float16* __restrict__ BT,
    const float* __restrict__ bias, float* __restrict__ Yf,
    _Float16* __restrict__ Yh, int M, int K, int Nn, int act, int out16) {
  __shared__ __align__(16) _Float16 As[2 * 64 * 64];  // [buf][m][k]
  __shared__ __align__(16) _Float16 Bs[2 * 64 * 64];  // [buf][n][k]
  const int tid = threadIdx.x;
  const int wave = tid >> 5, lane = tid & 31, cl = lane & 15;
  const int mi = wave & 3, ni = wave >> 2;
  const int n0 = blockIdx.x * 64;
  const int m0 = blockIdx.y * 64;

  // Per-thread staging slots: row sr and sr+32, 16 bytes at column s8.
  const int sr = tid >> 3;              // 0..31
  const int s8 = (tid & 7) * 8;         // f16 elements, 16B granules
  const uint32_t ga0 = (uint32_t)(((size_t)(m0 + sr) * K + s8) * 2);
  const uint32_t ga1 = ga0 + (uint32_t)(32 * K * 2);
  const uint32_t gb0 = (uint32_t)(((size_t)(n0 + sr) * K + s8) * 2);
  const uint32_t gb1 = gb0 + (uint32_t)(32 * K * 2);
  const uint32_t aL = led_lds_off(&As[sr * 64 + s8]);
  const uint32_t bL = led_lds_off(&Bs[sr * 64 + s8]);
  const uint32_t rowHalf = 32 * 64 * 2;   // bytes: rows +32
  const uint32_t bufStep = 64 * 64 * 2;   // bytes: buffer 1

  // Prologue: stream tile 0 into buffer 0.
  {
    LED_ASYNC_B128(aL, ga0, A);
    LED_ASYNC_B128(aL + rowHalf, ga1, A);
    LED_ASYNC_B128(bL, gb0, BT);
    LED_ASYNC_B128(bL + rowHalf, gb1, BT);
  }

  v8f acc[2] = {};
  for (int k0 = 0; k0 < K; k0 += 64) {
    const int cur = (k0 >> 6) & 1;
    __syncthreads();  // all waves finished reading buffer cur^1
    if (k0 + 64 < K) {
      const uint32_t kb = (uint32_t)((k0 + 64) * 2);
      const uint32_t lsel = (cur ^ 1) * bufStep;
      LED_ASYNC_B128(aL + lsel, ga0 + kb, A);
      LED_ASYNC_B128(aL + rowHalf + lsel, ga1 + kb, A);
      LED_ASYNC_B128(bL + lsel, gb0 + kb, BT);
      LED_ASYNC_B128(bL + rowHalf + lsel, gb1 + kb, BT);
      asm volatile("s_wait_asynccnt 0x4" ::: "memory");
    } else {
      asm volatile("s_wait_asynccnt 0x0" ::: "memory");
    }
    __syncthreads();  // buffer cur complete for all waves
    const _Float16* Ac = &As[cur * 64 * 64];
    const _Float16* Bc = &Bs[cur * 64 * 64];
#pragma unroll
    for (int kk = 0; kk < 64; kk += 32) {
      v16h a = led_fragA(&Ac[(mi * 16 + cl) * 64 + kk]);
#pragma unroll
      for (int s = 0; s < 2; ++s) {
        v16h b = led_fragB(&Bc[(ni * 32 + s * 16 + cl) * 64 + kk]);
        acc[s] = __builtin_amdgcn_wmma_f32_16x16x32_f16(
            false, a, false, b, (short)0, acc[s], false, false);
      }
    }
  }
  const int half8 = (lane >> 4) << 3;
#pragma unroll
  for (int s = 0; s < 2; ++s) {
    const int col = n0 + ni * 32 + s * 16 + cl;
    const float bvl = bias[col];
#pragma unroll
    for (int j = 0; j < 8; ++j) {
      float val = acc[s][j] + bvl;
      if (act == 1) val = 0.5f * val * (1.f + erff(val * 0.70710678f));
      else if (act == 2) val = tanhf(val);
      const int row = m0 + mi * 16 + j + half8;
      if (out16) Yh[(size_t)row * Nn + col] = (_Float16)val;
      else       Yf[(size_t)row * Nn + col] = val;
    }
  }
}

// ---------------------------------------------------------------------------
// Banded + global-column attention (flash-style). q/k/v f16, output f16.
// grid (C=32, H=12, 2 row-halves); block 256 = 8 waves, 16 q-rows per wave.
// ---------------------------------------------------------------------------
__global__ __launch_bounds__(256) void led_attn_local(
    const _Float16* __restrict__ q, const _Float16* __restrict__ k,
    const _Float16* __restrict__ v, const int* __restrict__ clss,
    const float* __restrict__ mask_src, const float* __restrict__ mask_cls,
    _Float16* __restrict__ attnout) {
  const int c = blockIdx.x, h = blockIdx.y, zhalf = blockIdx.z;
  const int tid = threadIdx.x;
  const int wave = tid >> 5, lane = tid & 31;
  const int cl = lane & 15;
  const int half8 = (lane >> 4) << 3;

  __shared__ __align__(16) _Float16 Qs[128 * 64];
  __shared__ __align__(16) _Float16 Ks[32 * 64];   // [key][d]
  __shared__ __align__(16) _Float16 Vts[64 * 32];  // [d][key]
  __shared__ __align__(16) _Float16 Ps[8 * 16 * 32];
  __shared__ float colmask[32];

  const int row0 = c * LED_W + zhalf * 128;
#pragma unroll
  for (int i = 0; i < 4; ++i) {
    const int idx = tid + i * 256;
    const int r = idx >> 3, d8 = (idx & 7) * 8;
    *(v8h*)&Qs[r * 64 + d8] =
        *(const v8h*)&q[(size_t)(row0 + r) * LED_D + h * LED_DH + d8];
  }
  __syncthreads();

  const v16h aq0 = led_fragA(&Qs[(wave * 16 + cl) * 64 + 0]);
  const v16h aq1 = led_fragA(&Qs[(wave * 16 + cl) * 64 + 32]);

  v8f acc[4] = {};
  float mrow[8], lrow[8];
#pragma unroll
  for (int j = 0; j < 8; ++j) { mrow[j] = -3.0e38f; lrow[j] = 0.f; }

  const int xbase = zhalf * 128 + wave * 16;
  const int skk = tid >> 3;            // staged key row
  const int sd8 = (tid & 7) * 8;       // d offset

  for (int t = 0; t < 28; ++t) {
    __syncthreads();
    if (tid < 32) {
      float cm;
      if (t < 4) {
        cm = (mask_cls[t * 32 + tid] > 0.f) ? 0.f : LED_NEG;
      } else {
        const int y = (t - 4) * 32 + tid;
        const int gpos = c * LED_W - LED_W + y;
        const bool inr = (gpos >= 0) && (gpos < LED_S);
        cm = (inr && mask_src[gpos] > 0.f) ? 0.f : LED_NEG;
      }
      colmask[tid] = cm;
    }
    {
      v8h kv8 = {}, vv8 = {};
      if (t < 4) {
        const int srow = clss[t * 32 + skk];
        kv8 = *(const v8h*)&k[(size_t)srow * LED_D + h * LED_DH + sd8];
        vv8 = *(const v8h*)&v[(size_t)srow * LED_D + h * LED_DH + sd8];
      } else {
        const int gpos = c * LED_W - LED_W + (t - 4) * 32 + skk;
        if (gpos >= 0 && gpos < LED_S) {
          kv8 = *(const v8h*)&k[(size_t)gpos * LED_D + h * LED_DH + sd8];
          vv8 = *(const v8h*)&v[(size_t)gpos * LED_D + h * LED_DH + sd8];
        }
      }
      *(v8h*)&Ks[skk * 64 + sd8] = kv8;
#pragma unroll
      for (int i = 0; i < 8; ++i) Vts[(sd8 + i) * 32 + skk] = vv8[i];
    }
    __syncthreads();

    v8f sc[2];
#pragma unroll
    for (int s = 0; s < 2; ++s) {
      const int col = s * 16 + cl;
      v16h b0 = led_fragB(&Ks[col * 64 + 0]);
      v16h b1 = led_fragB(&Ks[col * 64 + 32]);
      v8f z = {};
      z = __builtin_amdgcn_wmma_f32_16x16x32_f16(false, aq0, false, b0,
                                                 (short)0, z, false, false);
      z = __builtin_amdgcn_wmma_f32_16x16x32_f16(false, aq1, false, b1,
                                                 (short)0, z, false, false);
      sc[s] = z;
    }
#pragma unroll
    for (int s = 0; s < 2; ++s) {
      const int coltile = s * 16 + cl;
      const float cm = colmask[coltile];
#pragma unroll
      for (int j = 0; j < 8; ++j) {
        float val = sc[s][j] * LED_SCALE + cm;
        if (t >= 4) {
          const int x = xbase + j + half8;
          const int rel = (t - 4) * 32 + coltile - x;
          if (rel < 0 || rel > 2 * LED_W) val = LED_NEG;
        }
        sc[s][j] = val;
      }
    }
#pragma unroll
    for (int j = 0; j < 8; ++j) {
      float tm = fmaxf(sc[0][j], sc[1][j]);
#pragma unroll
      for (int off = 1; off < 16; off <<= 1)
        tm = fmaxf(tm, __shfl_xor(tm, off, 32));
      const float mnew = fmaxf(mrow[j], tm);
      const float fsc = __expf(mrow[j] - mnew);
      const float p0 = __expf(sc[0][j] - mnew);
      const float p1 = __expf(sc[1][j] - mnew);
      sc[0][j] = p0; sc[1][j] = p1;
      float rs = p0 + p1;
#pragma unroll
      for (int off = 1; off < 16; off <<= 1) rs += __shfl_xor(rs, off, 32);
      lrow[j] = lrow[j] * fsc + rs;
      mrow[j] = mnew;
#pragma unroll
      for (int o = 0; o < 4; ++o) acc[o][j] *= fsc;
    }
    _Float16* myP = &Ps[wave * 16 * 32];
#pragma unroll
    for (int s = 0; s < 2; ++s)
#pragma unroll
      for (int j = 0; j < 8; ++j)
        myP[(j + half8) * 32 + s * 16 + cl] = (_Float16)sc[s][j];
    asm volatile("s_wait_dscnt 0" ::: "memory");
    v16h ap = led_fragA(&myP[cl * 32]);
#pragma unroll
    for (int o = 0; o < 4; ++o) {
      v16h bv = led_fragB(&Vts[(o * 16 + cl) * 32]);
      acc[o] = __builtin_amdgcn_wmma_f32_16x16x32_f16(false, ap, false, bv,
                                                      (short)0, acc[o], false,
                                                      false);
    }
  }

#pragma unroll
  for (int j = 0; j < 8; ++j) {
    const float inv = 1.f / lrow[j];
    const int row = row0 + wave * 16 + j + half8;
    const size_t base = (size_t)row * LED_D + h * LED_DH;
    attnout[base + 0 + cl]  = (_Float16)(acc[0][j] * inv);
    attnout[base + 16 + cl] = (_Float16)(acc[1][j] * inv);
    attnout[base + 32 + cl] = (_Float16)(acc[2][j] * inv);
    attnout[base + 48 + cl] = (_Float16)(acc[3][j] * inv);
  }
}

// ---------------------------------------------------------------------------
// Global (cls-row) attention over all 8192 keys. grid = (H); block 256.
// ---------------------------------------------------------------------------
__global__ __launch_bounds__(256) void led_attn_global(
    const _Float16* __restrict__ qg, const _Float16* __restrict__ kga,
    const _Float16* __restrict__ vga, const float* __restrict__ mask_src,
    _Float16* __restrict__ og) {
  const int h = blockIdx.x;
  const int tid = threadIdx.x;
  const int wave = tid >> 5, lane = tid & 31;
  const int cl = lane & 15;
  const int half8 = (lane >> 4) << 3;

  __shared__ __align__(16) _Float16 Qs[128 * 64];
  __shared__ __align__(16) _Float16 Ks[32 * 64];
  __shared__ __align__(16) _Float16 Vts[64 * 32];
  __shared__ __align__(16) _Float16 Ps[8 * 16 * 32];
  __shared__ float colmask[32];

#pragma unroll
  for (int i = 0; i < 4; ++i) {
    const int idx = tid + i * 256;
    const int r = idx >> 3, d8 = (idx & 7) * 8;
    *(v8h*)&Qs[r * 64 + d8] =
        *(const v8h*)&qg[(size_t)r * LED_D + h * LED_DH + d8];
  }
  __syncthreads();
  const v16h aq0 = led_fragA(&Qs[(wave * 16 + cl) * 64 + 0]);
  const v16h aq1 = led_fragA(&Qs[(wave * 16 + cl) * 64 + 32]);

  v8f acc[4] = {};
  float mrow[8], lrow[8];
#pragma unroll
  for (int j = 0; j < 8; ++j) { mrow[j] = -3.0e38f; lrow[j] = 0.f; }

  const int skk = tid >> 3;
  const int sd8 = (tid & 7) * 8;

  for (int t = 0; t < LED_S / 32; ++t) {
    __syncthreads();
    if (tid < 32)
      colmask[tid] = (mask_src[t * 32 + tid] > 0.f) ? 0.f : LED_NEG;
    {
      const int gpos = t * 32 + skk;
      *(v8h*)&Ks[skk * 64 + sd8] =
          *(const v8h*)&kga[(size_t)gpos * LED_D + h * LED_DH + sd8];
      v8h vv8 = *(const v8h*)&vga[(size_t)gpos * LED_D + h * LED_DH + sd8];
#pragma unroll
      for (int i = 0; i < 8; ++i) Vts[(sd8 + i) * 32 + skk] = vv8[i];
    }
    __syncthreads();

    v8f sc[2];
#pragma unroll
    for (int s = 0; s < 2; ++s) {
      const int col = s * 16 + cl;
      v16h b0 = led_fragB(&Ks[col * 64 + 0]);
      v16h b1 = led_fragB(&Ks[col * 64 + 32]);
      v8f z = {};
      z = __builtin_amdgcn_wmma_f32_16x16x32_f16(false, aq0, false, b0,
                                                 (short)0, z, false, false);
      z = __builtin_amdgcn_wmma_f32_16x16x32_f16(false, aq1, false, b1,
                                                 (short)0, z, false, false);
      const float cm = colmask[col];
#pragma unroll
      for (int j = 0; j < 8; ++j) z[j] = z[j] * LED_SCALE + cm;
      sc[s] = z;
    }
#pragma unroll
    for (int j = 0; j < 8; ++j) {
      float tm = fmaxf(sc[0][j], sc[1][j]);
#pragma unroll
      for (int off = 1; off < 16; off <<= 1)
        tm = fmaxf(tm, __shfl_xor(tm, off, 32));
      const float mnew = fmaxf(mrow[j], tm);
      const float fsc = __expf(mrow[j] - mnew);
      const float p0 = __expf(sc[0][j] - mnew);
      const float p1 = __expf(sc[1][j] - mnew);
      sc[0][j] = p0; sc[1][j] = p1;
      float rs = p0 + p1;
#pragma unroll
      for (int off = 1; off < 16; off <<= 1) rs += __shfl_xor(rs, off, 32);
      lrow[j] = lrow[j] * fsc + rs;
      mrow[j] = mnew;
#pragma unroll
      for (int o = 0; o < 4; ++o) acc[o][j] *= fsc;
    }
    _Float16* myP = &Ps[wave * 16 * 32];
#pragma unroll
    for (int s = 0; s < 2; ++s)
#pragma unroll
      for (int j = 0; j < 8; ++j)
        myP[(j + half8) * 32 + s * 16 + cl] = (_Float16)sc[s][j];
    asm volatile("s_wait_dscnt 0" ::: "memory");
    v16h ap = led_fragA(&myP[cl * 32]);
#pragma unroll
    for (int o = 0; o < 4; ++o) {
      v16h bv = led_fragB(&Vts[(o * 16 + cl) * 32]);
      acc[o] = __builtin_amdgcn_wmma_f32_16x16x32_f16(false, ap, false, bv,
                                                      (short)0, acc[o], false,
                                                      false);
    }
  }
#pragma unroll
  for (int j = 0; j < 8; ++j) {
    const float inv = 1.f / lrow[j];
    const int n = wave * 16 + j + half8;
    const size_t base = (size_t)n * LED_D + h * LED_DH;
    og[base + 0 + cl]  = (_Float16)(acc[0][j] * inv);
    og[base + 16 + cl] = (_Float16)(acc[1][j] * inv);
    og[base + 32 + cl] = (_Float16)(acc[2][j] * inv);
    og[base + 48 + cl] = (_Float16)(acc[3][j] * inv);
  }
}

// ---------------------------------------------------------------------------
// Gather rows at cls indices (f32 source -> f16, optional per-row scale).
// ---------------------------------------------------------------------------
__global__ __launch_bounds__(256) void led_gather(
    const float* __restrict__ x, const int* __restrict__ clss,
    const float* __restrict__ rowscale, int use_scale,
    _Float16* __restrict__ out) {
  const int n = blockIdx.x;
  const int tid = threadIdx.x;
  const int srow = clss[n];
  const float sc = use_scale ? rowscale[n] : 1.f;
#pragma unroll
  for (int i = 0; i < 3; ++i) {
    const int c = tid + i * 256;
    out[(size_t)n * LED_D + c] = (_Float16)(x[(size_t)srow * LED_D + c] * sc);
  }
}

__global__ __launch_bounds__(256) void led_scatter(
    const _Float16* __restrict__ og, const int* __restrict__ clss,
    _Float16* __restrict__ attnout) {
  const int n = blockIdx.x;
  const int tid = threadIdx.x;
  const int drow = clss[n];
#pragma unroll
  for (int i = 0; i < 3; ++i) {
    const int c = tid + i * 256;
    attnout[(size_t)drow * LED_D + c] = og[(size_t)n * LED_D + c];
  }
}

// ---------------------------------------------------------------------------
// Head: out = [sigmoid(h1 @ oW + ob) * mask | mask]
// ---------------------------------------------------------------------------
__global__ __launch_bounds__(128) void led_head(
    const float* __restrict__ h1, const float* __restrict__ oW,
    const float* __restrict__ ob, const float* __restrict__ mask_cls,
    float* __restrict__ out) {
  const int n = threadIdx.x;
  float s = 0.f;
  for (int d = 0; d < LED_D; d += 4) {
    const float4 hv = *(const float4*)&h1[(size_t)n * LED_D + d];
    const float4 wv = *(const float4*)&oW[d];
    s += hv.x * wv.x + hv.y * wv.y + hv.z * wv.z + hv.w * wv.w;
  }
  s += ob[0];
  const float sig = 1.f / (1.f + __expf(-s));
  out[n] = sig * mask_cls[n];
  out[LED_N + n] = mask_cls[n];
}

// ---------------------------------------------------------------------------
// Host orchestration
// ---------------------------------------------------------------------------
extern "C" void kernel_launch(void* const* d_in, const int* in_sizes, int n_in,
                              void* d_out, int out_size, void* d_ws,
                              size_t ws_size, hipStream_t stream) {
  (void)in_sizes; (void)n_in; (void)out_size; (void)ws_size;

  const int*   src      = (const int*)d_in[0];
  const int*   clss     = (const int*)d_in[3];
  const float* mask_src = (const float*)d_in[4];
  const float* mask_cls = (const float*)d_in[5];
  const float* emb   = (const float*)d_in[6];
  const float* pos   = (const float*)d_in[7];
  const float* ln_es = (const float*)d_in[8];
  const float* ln_eb = (const float*)d_in[9];
  const float* Wq  = (const float*)d_in[10];
  const float* Wk  = (const float*)d_in[11];
  const float* Wv  = (const float*)d_in[12];
  const float* Wo  = (const float*)d_in[13];
  const float* Wqg = (const float*)d_in[14];
  const float* Wkg = (const float*)d_in[15];
  const float* Wvg = (const float*)d_in[16];
  const float* bq  = (const float*)d_in[17];
  const float* bk  = (const float*)d_in[18];
  const float* bv  = (const float*)d_in[19];
  const float* bo  = (const float*)d_in[20];
  const float* bqg = (const float*)d_in[21];
  const float* bkg = (const float*)d_in[22];
  const float* bvg = (const float*)d_in[23];
  const float* ln1s = (const float*)d_in[24];
  const float* ln1b = (const float*)d_in[25];
  const float* W1 = (const float*)d_in[26];
  const float* b1 = (const float*)d_in[27];
  const float* W2 = (const float*)d_in[28];
  const float* b2 = (const float*)d_in[29];
  const float* ln2s = (const float*)d_in[30];
  const float* ln2b = (const float*)d_in[31];
  const float* hW = (const float*)d_in[32];
  const float* hb = (const float*)d_in[33];
  const float* oW = (const float*)d_in[34];
  const float* ob = (const float*)d_in[35];

  float* ws = (float*)d_ws;
  const size_t SD = (size_t)LED_S * LED_D;
  const size_t ND = (size_t)LED_N * LED_D;
  const size_t DD = (size_t)LED_D * LED_D;
  const size_t DF = (size_t)LED_D * LED_FF;
  const size_t SF = (size_t)LED_S * LED_FF;

  // f32 region
  float* X  = ws;
  float* HB = X + SD;
  float* H1 = HB + SD;
  // f16 region
  _Float16* hbase = (_Float16*)(H1 + ND);
  _Float16* Xh  = hbase;
  _Float16* Qh  = Xh + SD;
  _Float16* Kh  = Qh + SD;
  _Float16* Vh  = Kh + SD;
  _Float16* KGh = Vh + SD;
  _Float16* VGh = KGh + SD;
  _Float16* AOh = VGh + SD;
  _Float16* F1h = AOh + SD;
  _Float16* QGh = F1h + SF;
  _Float16* OGh = QGh + ND;
  _Float16* XGh = OGh + ND;
  _Float16* SVh = XGh + ND;
  _Float16* WqT  = SVh + ND;
  _Float16* WkT  = WqT + 2 * DD;
  _Float16* WvT  = WkT + 2 * DD;
  _Float16* WoT  = WvT + 2 * DD;
  _Float16* WqgT = WoT + 2 * DD;
  _Float16* WkgT = WqgT + 2 * DD;
  _Float16* WvgT = WkgT + 2 * DD;
  _Float16* W1T  = WvgT + 2 * DD;   // [FF][D] per layer
  _Float16* W2T  = W1T + 2 * DF;    // [D][FF] per layer
  _Float16* hWT  = W2T + 2 * DF;

  const dim3 blk(256);
  const dim3 tDD(LED_D / 32, LED_D / 32);
  const dim3 tDF(LED_FF / 32, LED_D / 32);   // W1: K=D rows, N=FF cols
  const dim3 tFD(LED_D / 32, LED_FF / 32);   // W2: K=FF rows, N=D cols
  const dim3 gD(LED_D / 64, LED_S / 64);
  const dim3 gF(LED_FF / 64, LED_S / 64);
  const dim3 gDn(LED_D / 64, LED_N / 64);

  // ---- weight convert + transpose (deterministic, every call)
  for (int l = 0; l < 2; ++l) {
    led_w_to_f16t<<<tDD, blk, 0, stream>>>(Wq + l * DD, WqT + l * DD, LED_D, LED_D);
    led_w_to_f16t<<<tDD, blk, 0, stream>>>(Wk + l * DD, WkT + l * DD, LED_D, LED_D);
    led_w_to_f16t<<<tDD, blk, 0, stream>>>(Wv + l * DD, WvT + l * DD, LED_D, LED_D);
    led_w_to_f16t<<<tDD, blk, 0, stream>>>(Wo + l * DD, WoT + l * DD, LED_D, LED_D);
    led_w_to_f16t<<<tDD, blk, 0, stream>>>(Wqg + l * DD, WqgT + l * DD, LED_D, LED_D);
    led_w_to_f16t<<<tDD, blk, 0, stream>>>(Wkg + l * DD, WkgT + l * DD, LED_D, LED_D);
    led_w_to_f16t<<<tDD, blk, 0, stream>>>(Wvg + l * DD, WvgT + l * DD, LED_D, LED_D);
    led_w_to_f16t<<<tDF, blk, 0, stream>>>(W1 + l * DF, W1T + l * DF, LED_D, LED_FF);
    led_w_to_f16t<<<tFD, blk, 0, stream>>>(W2 + l * DF, W2T + l * DF, LED_FF, LED_D);
  }
  led_w_to_f16t<<<tDD, blk, 0, stream>>>(hW, hWT, LED_D, LED_D);

  led_embed_ln<<<LED_S, blk, 0, stream>>>(src, emb, pos, ln_es, ln_eb, X, Xh);

  for (int l = 0; l < 2; ++l) {
    const size_t oDD = (size_t)l * DD;
    const size_t oD  = (size_t)l * LED_D;
    const size_t oDF = (size_t)l * DF;
    const size_t oF  = (size_t)l * LED_FF;

    led_gemm_wmma<<<gD, blk, 0, stream>>>(Xh, WqT + oDD, bq + oD, nullptr, Qh,
                                          LED_S, LED_D, LED_D, 0, 1);
    led_gemm_wmma<<<gD, blk, 0, stream>>>(Xh, WkT + oDD, bk + oD, nullptr, Kh,
                                          LED_S, LED_D, LED_D, 0, 1);
    led_gemm_wmma<<<gD, blk, 0, stream>>>(Xh, WvT + oDD, bv + oD, nullptr, Vh,
                                          LED_S, LED_D, LED_D, 0, 1);
    led_gemm_wmma<<<gD, blk, 0, stream>>>(Xh, WkgT + oDD, bkg + oD, nullptr,
                                          KGh, LED_S, LED_D, LED_D, 0, 1);
    led_gemm_wmma<<<gD, blk, 0, stream>>>(Xh, WvgT + oDD, bvg + oD, nullptr,
                                          VGh, LED_S, LED_D, LED_D, 0, 1);
    led_gather<<<LED_N, blk, 0, stream>>>(X, clss, mask_cls, 0, XGh);
    led_gemm_wmma<<<gDn, blk, 0, stream>>>(XGh, WqgT + oDD, bqg + oD, nullptr,
                                           QGh, LED_N, LED_D, LED_D, 0, 1);

    led_attn_local<<<dim3(LED_C, LED_H, 2), blk, 0, stream>>>(
        Qh, Kh, Vh, clss, mask_src, mask_cls, AOh);
    led_attn_global<<<dim3(LED_H), blk, 0, stream>>>(QGh, KGh, VGh, mask_src,
                                                     OGh);
    led_scatter<<<LED_N, blk, 0, stream>>>(OGh, clss, AOh);

    led_gemm_wmma<<<gD, blk, 0, stream>>>(AOh, WoT + oDD, bo + oD, HB, nullptr,
                                          LED_S, LED_D, LED_D, 0, 0);
    led_add_ln<<<LED_S, blk, 0, stream>>>(X, HB, ln1s + oD, ln1b + oD, X, Xh);

    led_gemm_wmma<<<gF, blk, 0, stream>>>(Xh, W1T + oDF, b1 + oF, nullptr, F1h,
                                          LED_S, LED_D, LED_FF, 1, 1);
    led_gemm_wmma<<<gD, blk, 0, stream>>>(F1h, W2T + oDF, b2 + oD, HB, nullptr,
                                          LED_S, LED_FF, LED_D, 0, 0);
    led_add_ln<<<LED_S, blk, 0, stream>>>(X, HB, ln2s + oD, ln2b + oD, X, Xh);
  }

  led_gather<<<LED_N, blk, 0, stream>>>(X, clss, mask_cls, 1, SVh);
  led_gemm_wmma<<<gDn, blk, 0, stream>>>(SVh, hWT, hb, H1, nullptr,
                                         LED_N, LED_D, LED_D, 2, 0);
  led_head<<<1, 128, 0, stream>>>(H1, oW, ob, mask_cls, (float*)d_out);
}
